// MultiHeadCrossAttention_78726750536492
// MI455X (gfx1250) — matrix operand order repaired and checked by reading, not compiled, not run on hardware
//
#include <hip/hip_runtime.h>
#include <hip/hip_bf16.h>

typedef __bf16 bf16_t;
typedef __attribute__((ext_vector_type(16))) __bf16 v16bf;
typedef __attribute__((ext_vector_type(8)))  __bf16 v8bf;
typedef __attribute__((ext_vector_type(8)))  float  v8f;

#define WMMA_BF16(A_, B_, C_) \
  __builtin_amdgcn_wmma_f32_16x16x32_bf16(false, (A_), false, (B_), (short)0, (C_), false, false)

static __device__ __forceinline__ v16bf cat8(v8bf lo, v8bf hi) {
  return __builtin_shufflevector(lo, hi, 0,1,2,3,4,5,6,7,8,9,10,11,12,13,14,15);
}

// ---------------------------------------------------------------------------
// gfx1250 async copy: global memory -> LDS, 16B per lane, tracked by ASYNCcnt
// ---------------------------------------------------------------------------
static __device__ __forceinline__ void async_copy_b128(void* lds_ptr, const void* gptr) {
  unsigned int lds_off =
      (unsigned int)(unsigned long long)(__attribute__((address_space(3))) void*)lds_ptr;
  asm volatile("global_load_async_to_lds_b128 %0, %1, off"
               :: "v"(lds_off), "v"(gptr)
               : "memory");
}
static __device__ __forceinline__ void wait_async0() {
  asm volatile("s_wait_asynccnt 0" ::: "memory");
}

// ---------------------------------------------------------------------------
// fp32 -> bf16 grid-stride conversion
// ---------------------------------------------------------------------------
__global__ __launch_bounds__(256)
void cvt_f32_bf16(const float* __restrict__ src, bf16_t* __restrict__ dst, long n) {
  long i = (long)blockIdx.x * blockDim.x + threadIdx.x;
  long stride = (long)gridDim.x * blockDim.x;
  for (; i < n; i += stride) dst[i] = (bf16_t)src[i];
}

// ---------------------------------------------------------------------------
// Generic GEMM:  out = act(A @ W^T + bias) [+ resid]
//   A : M x K bf16 row-major, W : N x K bf16 row-major (torch Linear layout)
//   block tile 128x128, 8 waves (2x4), wave tile 64x32, K-slab 32
//   double-buffered LDS, async global->LDS staging overlapped with WMMA
// ---------------------------------------------------------------------------
#define KP 40   // LDS K stride in elements (80 bytes, 16B aligned)

__global__ __launch_bounds__(256)
void gemm_xwT(const bf16_t* __restrict__ A, const bf16_t* __restrict__ W,
              const float* __restrict__ bias, const float* __restrict__ resid,
              float* __restrict__ outF, bf16_t* __restrict__ outB,
              int M, int N, int K, int act)
{
  __shared__ bf16_t As[2][128][KP];
  __shared__ bf16_t Ws[2][128][KP];

  const int tid  = threadIdx.x;
  const int lane = tid & 31;
  const int wave = tid >> 5;
  const int wm   = wave >> 2;       // 0..1
  const int wn   = wave & 3;        // 0..3
  const int m0   = blockIdx.y * 128;
  const int n0   = blockIdx.x * 128;

  const v8f zacc = {};
  v8f acc[4][2];
#pragma unroll
  for (int i = 0; i < 4; ++i)
#pragma unroll
    for (int j = 0; j < 2; ++j) acc[i][j] = zacc;

  const int lr = tid >> 2;          // 0..63  (load row)
  const int lc = (tid & 3) << 3;    // 0,8,16,24

  const int hrow = lane & 15;
  const int kba  = (lane >> 4) << 3;   // 0 / 8   (unused; A and B frags use kbb)
  const int kbb  = (lane >> 4) << 4;   // 0 / 16  (B frag)

  // stage K-slab `kk` into buffer `buf` (async, 4 x b128 per thread)
  auto stage = [&](int buf, int kk) {
#pragma unroll
    for (int p = 0; p < 2; ++p) {
      int row = lr + p * 64;
      async_copy_b128(&As[buf][row][lc], A + (size_t)(m0 + row) * K + kk + lc);
      async_copy_b128(&Ws[buf][row][lc], W + (size_t)(n0 + row) * K + kk + lc);
    }
  };

  stage(0, 0);
  wait_async0();
  __syncthreads();

  for (int kk = 0; kk < K; kk += 32) {
    const int cur = (kk >> 5) & 1;
    if (kk + 32 < K) stage(cur ^ 1, kk + 32);   // prefetch next slab during compute

    v16bf af[4], bfr[2];
#pragma unroll
    for (int mt = 0; mt < 4; ++mt) {
      const bf16_t* ap = &As[cur][wm * 64 + mt * 16 + hrow][0];
      af[mt] = cat8(*(const v8bf*)&ap[kbb], *(const v8bf*)&ap[kbb + 8]);
    }
#pragma unroll
    for (int nt = 0; nt < 2; ++nt) {
      const bf16_t* bp = &Ws[cur][wn * 32 + nt * 16 + hrow][0];
      bfr[nt] = cat8(*(const v8bf*)&bp[kbb], *(const v8bf*)&bp[kbb + 8]);
    }
#pragma unroll
    for (int mt = 0; mt < 4; ++mt)
#pragma unroll
      for (int nt = 0; nt < 2; ++nt)
        acc[mt][nt] = WMMA_BF16(af[mt], bfr[nt], acc[mt][nt]);

    wait_async0();       // next-slab async fills complete
    __syncthreads();     // all waves done reading `cur` before it is refilled
  }

  const int crow = (lane >> 4) << 3;   // 0 / 8
  const int ccol = lane & 15;
#pragma unroll
  for (int mt = 0; mt < 4; ++mt) {
#pragma unroll
    for (int nt = 0; nt < 2; ++nt) {
      int n = n0 + wn * 32 + nt * 16 + ccol;
      float bv = bias ? bias[n] : 0.0f;
#pragma unroll
      for (int r = 0; r < 8; ++r) {
        int m = m0 + wm * 64 + mt * 16 + crow + r;
        float v = acc[mt][nt][r] + bv;
        if (act == 1) v = 0.5f * v * (1.0f + erff(v * 0.70710678118654752f)); // exact GELU
        size_t idx = (size_t)m * N + n;
        if (resid) v += resid[idx];
        if (outF) outF[idx] = v;
        if (outB) outB[idx] = (bf16_t)v;
      }
    }
  }
}

// ---------------------------------------------------------------------------
// Attention scores:  scores[b,h,t,s] = (q . k) * 1/sqrt(64), masked
//   per block: one (b,h), 128 rows of T, 128 cols of S; K = Dh = 64 in LDS
// ---------------------------------------------------------------------------
__global__ __launch_bounds__(256)
void attn_scores(const bf16_t* __restrict__ Q, const bf16_t* __restrict__ Km,
                 const unsigned char* __restrict__ kpm, float* __restrict__ attnW,
                 int T, int S, int E, int Hh)
{
  const int bh = blockIdx.z;
  const int b  = bh / Hh;
  const int h  = bh % Hh;
  const int t0 = blockIdx.y * 128;
  const int s0 = blockIdx.x * 128;

  const bf16_t* Qb = Q  + (size_t)b * T * E + (size_t)h * 64;
  const bf16_t* Kb = Km + (size_t)b * S * E + (size_t)h * 64;

  __shared__ bf16_t Qs[128][72];   // 64 + pad8 -> 144B rows (16B aligned)
  __shared__ bf16_t Ks[128][72];

  const int tid  = threadIdx.x;
  const int lane = tid & 31;
  const int wave = tid >> 5;
  const int wm   = wave >> 2;      // 0..1
  const int wn   = wave & 3;       // 0..3

#pragma unroll
  for (int p = 0; p < 4; ++p) {
    int row = p * 32 + (tid >> 3);
    int c   = (tid & 7) << 3;
    async_copy_b128(&Qs[row][c], Qb + (size_t)(t0 + row) * E + c);
    async_copy_b128(&Ks[row][c], Kb + (size_t)(s0 + row) * E + c);
  }
  wait_async0();
  __syncthreads();

  const v8f zacc = {};
  v8f acc[4][2];
#pragma unroll
  for (int i = 0; i < 4; ++i)
#pragma unroll
    for (int j = 0; j < 2; ++j) acc[i][j] = zacc;

  const int hrow = lane & 15;
  const int kba  = (lane >> 4) << 3;
  const int kbb  = (lane >> 4) << 4;

#pragma unroll
  for (int ks = 0; ks < 64; ks += 32) {
    v16bf af[4], bfr[2];
#pragma unroll
    for (int mt = 0; mt < 4; ++mt) {
      const bf16_t* ap = &Qs[wm * 64 + mt * 16 + hrow][ks];
      af[mt] = cat8(*(const v8bf*)&ap[kbb], *(const v8bf*)&ap[kbb + 8]);
    }
#pragma unroll
    for (int nt = 0; nt < 2; ++nt) {
      const bf16_t* bp = &Ks[wn * 32 + nt * 16 + hrow][ks];
      bfr[nt] = cat8(*(const v8bf*)&bp[kbb], *(const v8bf*)&bp[kbb + 8]);
    }
#pragma unroll
    for (int mt = 0; mt < 4; ++mt)
#pragma unroll
      for (int nt = 0; nt < 2; ++nt)
        acc[mt][nt] = WMMA_BF16(af[mt], bfr[nt], acc[mt][nt]);
  }

  const int crow = (lane >> 4) << 3;
  const int ccol = lane & 15;
  float* base = attnW + (size_t)bh * T * S;
#pragma unroll
  for (int mt = 0; mt < 4; ++mt) {
#pragma unroll
    for (int nt = 0; nt < 2; ++nt) {
      int s = s0 + wn * 32 + nt * 16 + ccol;
      bool valid = kpm[(size_t)b * S + s] != 0;
#pragma unroll
      for (int r = 0; r < 8; ++r) {
        int t = t0 + wm * 64 + mt * 16 + crow + r;
        float v = acc[mt][nt][r] * 0.125f;             // 1/sqrt(64)
        base[(size_t)t * S + s] = valid ? v : -1.0e9f;
      }
    }
  }
}

// ---------------------------------------------------------------------------
// Row softmax in place over S columns (one block per row)
// ---------------------------------------------------------------------------
__global__ __launch_bounds__(256)
void softmax_rows(float* __restrict__ attnW, int S)
{
  float* p = attnW + (size_t)blockIdx.x * S;
  __shared__ float red[256];
  const int tid = threadIdx.x;

  float m = -3.0e38f;
  for (int i = tid; i < S; i += 256) m = fmaxf(m, p[i]);
  red[tid] = m; __syncthreads();
  for (int s = 128; s > 0; s >>= 1) {
    if (tid < s) red[tid] = fmaxf(red[tid], red[tid + s]);
    __syncthreads();
  }
  const float rowmax = red[0];
  __syncthreads();

  float sum = 0.0f;
  for (int i = tid; i < S; i += 256) {
    float e = __expf(p[i] - rowmax);
    p[i] = e;
    sum += e;
  }
  red[tid] = sum; __syncthreads();
  for (int s = 128; s > 0; s >>= 1) {
    if (tid < s) red[tid] += red[tid + s];
    __syncthreads();
  }
  const float inv = 1.0f / red[0];
  for (int i = tid; i < S; i += 256) p[i] *= inv;
}

// ---------------------------------------------------------------------------
// ctx[b,h,t,d] = sum_s attn[b,h,t,s] * v[b,s,h*64+d]
//   per block: one (b,h), 128 rows of T, full N=64; attn f32->bf16 on the fly
//   8 waves (4x2), wave tile 32x32; V tile staged via async copy
// ---------------------------------------------------------------------------
__global__ __launch_bounds__(256)
void attn_ctx(const float* __restrict__ attnW, const bf16_t* __restrict__ V,
              bf16_t* __restrict__ ctx, int T, int S, int E, int Hh)
{
  const int bh = blockIdx.y;
  const int b  = bh / Hh;
  const int h  = bh % Hh;
  const int t0 = blockIdx.x * 128;

  const float*  Ab = attnW + (size_t)bh * T * S;
  const bf16_t* Vb = V + (size_t)b * S * E + (size_t)h * 64;

  __shared__ bf16_t As[128][KP];
  __shared__ bf16_t Vs[32][72];

  const int tid  = threadIdx.x;
  const int lane = tid & 31;
  const int wave = tid >> 5;
  const int wm   = wave >> 1;      // 0..3
  const int wn   = wave & 1;       // 0..1

  const v8f zacc = {};
  v8f acc[2][2];
#pragma unroll
  for (int i = 0; i < 2; ++i)
#pragma unroll
    for (int j = 0; j < 2; ++j) acc[i][j] = zacc;

  const int hrow = lane & 15;
  const int kba  = (lane >> 4) << 3;
  const int kbb  = (lane >> 4) << 4;

  for (int kk = 0; kk < S; kk += 32) {
    __syncthreads();
    {
      // V tile: async global->LDS
      int vrow = tid >> 3;          // 0..31
      int vc   = (tid & 7) << 3;
      async_copy_b128(&Vs[vrow][vc], Vb + (size_t)(kk + vrow) * E + vc);
      // attn tile: fp32 -> bf16 via VALU
      int row = tid >> 1;
      int c0  = (tid & 1) << 4;
      const float* src = Ab + (size_t)(t0 + row) * S + kk + c0;
#pragma unroll
      for (int e = 0; e < 16; ++e) As[row][c0 + e] = (bf16_t)src[e];
    }
    wait_async0();
    __syncthreads();

    v16bf af[2], bfr[2];
#pragma unroll
    for (int mt = 0; mt < 2; ++mt) {
      const bf16_t* ap = &As[wm * 32 + mt * 16 + hrow][0];
      af[mt] = cat8(*(const v8bf*)&ap[kbb], *(const v8bf*)&ap[kbb + 8]);
    }
#pragma unroll
    for (int nt = 0; nt < 2; ++nt) {
      int ncol = wn * 32 + nt * 16 + hrow;
      v16bf bb;
#pragma unroll
      for (int e = 0; e < 16; ++e) bb[e] = Vs[kbb + e][ncol];
      bfr[nt] = bb;
    }
#pragma unroll
    for (int mt = 0; mt < 2; ++mt)
#pragma unroll
      for (int nt = 0; nt < 2; ++nt)
        acc[mt][nt] = WMMA_BF16(af[mt], bfr[nt], acc[mt][nt]);
  }

  const int crow = (lane >> 4) << 3;
  const int ccol = lane & 15;
#pragma unroll
  for (int mt = 0; mt < 2; ++mt) {
#pragma unroll
    for (int nt = 0; nt < 2; ++nt) {
      int d = wn * 32 + nt * 16 + ccol;
#pragma unroll
      for (int r = 0; r < 8; ++r) {
        int t = t0 + wm * 32 + mt * 16 + crow + r;
        ctx[((size_t)b * T + t) * E + (size_t)h * 64 + d] = (bf16_t)acc[mt][nt][r];
      }
    }
  }
}

// ---------------------------------------------------------------------------
// LayerNorm over last dim E (one block per row)
// ---------------------------------------------------------------------------
__global__ __launch_bounds__(256)
void layernorm_rows(const float* __restrict__ X, const float* __restrict__ g,
                    const float* __restrict__ bta, float* __restrict__ out, int E)
{
  const float* x = X + (size_t)blockIdx.x * E;
  float* o = out + (size_t)blockIdx.x * E;
  __shared__ float red[256];
  const int tid = threadIdx.x;

  float s = 0.0f;
  for (int i = tid; i < E; i += 256) s += x[i];
  red[tid] = s; __syncthreads();
  for (int k = 128; k > 0; k >>= 1) {
    if (tid < k) red[tid] += red[tid + k];
    __syncthreads();
  }
  const float mu = red[0] / (float)E;
  __syncthreads();

  float v = 0.0f;
  for (int i = tid; i < E; i += 256) { float d = x[i] - mu; v += d * d; }
  red[tid] = v; __syncthreads();
  for (int k = 128; k > 0; k >>= 1) {
    if (tid < k) red[tid] += red[tid + k];
    __syncthreads();
  }
  const float rstd = rsqrtf(red[0] / (float)E + 1e-5f);

  for (int i = tid; i < E; i += 256)
    o[i] = (x[i] - mu) * rstd * g[i] + bta[i];
}

// ---------------------------------------------------------------------------
extern "C" void kernel_launch(void* const* d_in, const int* in_sizes, int n_in,
                              void* d_out, int out_size, void* d_ws, size_t ws_size,
                              hipStream_t stream) {
  (void)in_sizes; (void)n_in; (void)out_size; (void)ws_size;

  const float* query      = (const float*)d_in[0];
  const float* key        = (const float*)d_in[1];
  const float* value      = (const float*)d_in[2];
  const unsigned char* kpm = (const unsigned char*)d_in[3];
  const float* in_proj_w  = (const float*)d_in[4];
  const float* in_proj_b  = (const float*)d_in[5];
  const float* out_w      = (const float*)d_in[6];
  const float* out_b      = (const float*)d_in[7];
  const float* ffn_w1     = (const float*)d_in[8];
  const float* ffn_b1     = (const float*)d_in[9];
  const float* ffn_w2     = (const float*)d_in[10];
  const float* ffn_b2     = (const float*)d_in[11];
  const float* ln_g       = (const float*)d_in[12];
  const float* ln_b       = (const float*)d_in[13];

  const int B = 8, T = 512, S = 1024, E = 1024, Hh = 16, FH = 4096;
  const int MT = B * T;   // 4096 query rows
  const int MS = B * S;   // 8192 key/value rows

  float* outp  = (float*)d_out;                  // (B,T,E)
  float* attnW = outp + (size_t)MT * E;          // (B,H,T,S)

  // workspace layout (256B aligned slices)
  char* ws = (char*)d_ws;
  size_t off = 0;
  auto alloc = [&](size_t bytes) -> void* {
    void* p = ws + off;
    off += (bytes + 255) & ~(size_t)255;
    return p;
  };
  bf16_t* qry_bf = (bf16_t*)alloc((size_t)MT * E * 2);
  bf16_t* key_bf = (bf16_t*)alloc((size_t)MS * E * 2);
  bf16_t* val_bf = (bf16_t*)alloc((size_t)MS * E * 2);
  bf16_t* win_bf = (bf16_t*)alloc((size_t)3 * E * E * 2);
  bf16_t* wo_bf  = (bf16_t*)alloc((size_t)E * E * 2);
  bf16_t* w1_bf  = (bf16_t*)alloc((size_t)FH * E * 2);
  bf16_t* w2_bf  = (bf16_t*)alloc((size_t)E * FH * 2);
  bf16_t* q_bf   = (bf16_t*)alloc((size_t)MT * E * 2);
  bf16_t* k_bf   = (bf16_t*)alloc((size_t)MS * E * 2);
  bf16_t* v_bf   = (bf16_t*)alloc((size_t)MS * E * 2);
  bf16_t* ctx_bf = (bf16_t*)alloc((size_t)MT * E * 2);
  bf16_t* ao_bf  = (bf16_t*)alloc((size_t)MT * E * 2);
  bf16_t* h_bf   = (bf16_t*)alloc((size_t)MT * FH * 2);
  float*  ao_f32 = (float*)alloc((size_t)MT * E * 4);
  float*  x_f32  = (float*)alloc((size_t)MT * E * 4);

  // --- 1) convert inputs + weights to bf16 ---
  auto cvt = [&](const float* s, bf16_t* d, long n) {
    int blocks = (int)((n + 2047) / 2048);
    cvt_f32_bf16<<<blocks, 256, 0, stream>>>(s, d, n);
  };
  cvt(query,     qry_bf, (long)MT * E);
  cvt(key,       key_bf, (long)MS * E);
  cvt(value,     val_bf, (long)MS * E);
  cvt(in_proj_w, win_bf, (long)3 * E * E);
  cvt(out_w,     wo_bf,  (long)E * E);
  cvt(ffn_w1,    w1_bf,  (long)FH * E);
  cvt(ffn_w2,    w2_bf,  (long)E * FH);

  // --- 2) QKV projections ---
  gemm_xwT<<<dim3(E / 128, MT / 128), 256, 0, stream>>>(
      qry_bf, win_bf,                   in_proj_b,         nullptr, nullptr, q_bf, MT, E, E, 0);
  gemm_xwT<<<dim3(E / 128, MS / 128), 256, 0, stream>>>(
      key_bf, win_bf + (size_t)E * E,   in_proj_b + E,     nullptr, nullptr, k_bf, MS, E, E, 0);
  gemm_xwT<<<dim3(E / 128, MS / 128), 256, 0, stream>>>(
      val_bf, win_bf + (size_t)2 * E * E, in_proj_b + 2 * E, nullptr, nullptr, v_bf, MS, E, E, 0);

  // --- 3) attention ---
  attn_scores<<<dim3(S / 128, T / 128, B * Hh), 256, 0, stream>>>(
      q_bf, k_bf, kpm, attnW, T, S, E, Hh);
  softmax_rows<<<B * Hh * T, 256, 0, stream>>>(attnW, S);
  attn_ctx<<<dim3(T / 128, B * Hh), 256, 0, stream>>>(
      attnW, v_bf, ctx_bf, T, S, E, Hh);

  // --- 4) output projection (keep f32 for residual, bf16 for FFN input) ---
  gemm_xwT<<<dim3(E / 128, MT / 128), 256, 0, stream>>>(
      ctx_bf, wo_bf, out_b, nullptr, ao_f32, ao_bf, MT, E, E, 0);

  // --- 5) FFN ---
  gemm_xwT<<<dim3(FH / 128, MT / 128), 256, 0, stream>>>(
      ao_bf, w1_bf, ffn_b1, nullptr, nullptr, h_bf, MT, FH, E, 1 /*gelu*/);
  gemm_xwT<<<dim3(E / 128, MT / 128), 256, 0, stream>>>(
      h_bf, w2_bf, ffn_b2, ao_f32 /*residual*/, x_f32, nullptr, MT, E, FH, 0);

  // --- 6) LayerNorm -> output ---
  layernorm_rows<<<MT, 256, 0, stream>>>(x_f32, ln_g, ln_b, outp, E);
}
